// P2SP_41394894799780
// MI455X (gfx1250) — compile-verified, matrix operand order checked by reading, not run on hardware
//
#include <hip/hip_runtime.h>
#include <hip/hip_bf16.h>

typedef __attribute__((ext_vector_type(16))) _Float16 v16h;
typedef __attribute__((ext_vector_type(8)))  _Float16 v8h;
typedef __attribute__((ext_vector_type(8)))  float    v8f;

// K-dim packing for 16-bit A/B fragments of V_WMMA_*_16X16X32 (ISA 7.12.2).
// j=0..7  -> K = half*8 + j        (contiguous)
// j=8..15 -> K = 16 + half*8 + j-8 (contiguous)
__device__ __forceinline__ int kpack(int j, int half) {
  return ((j >> 3) << 4) | (half << 3) | (((j >> 1) & 3) << 1) | (j & 1);
}

__device__ __forceinline__ v16h frag16(const _Float16* __restrict__ p) {
  v8h lo = *(const v8h*)p;        // K = base + half*8 + 0..7
  v8h hi = *(const v8h*)(p + 16); // K = base + 16 + half*8 + 0..7
  return __builtin_shufflevector(lo, hi, 0, 1, 2, 3, 4, 5, 6, 7,
                                 8, 9, 10, 11, 12, 13, 14, 15);
}

// ---------------------------------------------------------------------------
// Pack fp32 weight matrix (Cout x Cin row major) into f16 A-fragment order.
// ---------------------------------------------------------------------------
__global__ void packw_k(const float* __restrict__ W, _Float16* __restrict__ Wf,
                        int Cout, int Cin) {
  const int KT = Cin >> 5;
  const int total = Cout * Cin;
  int tid = blockIdx.x * blockDim.x + threadIdx.x;
  if (tid >= total) return;
  int j    = tid & 15;
  int lane = (tid >> 4) & 31;
  int rest = tid >> 9;
  int kt = rest % KT;
  int ot = rest / KT;
  int o = ot * 16 + (lane & 15);
  int k = kt * 32 + kpack(j, lane >> 4);
  Wf[tid] = (_Float16)W[(long)o * Cin + k];
}

// ---------------------------------------------------------------------------
// Transpose activations NCHW fp32 -> [b*hw][C] f16 (GEMM B operand layout).
// ---------------------------------------------------------------------------
__global__ void pack_act_k(const float* __restrict__ X, _Float16* __restrict__ Xt,
                           int C, int HW, int total) {
  int t = blockIdx.x * blockDim.x + threadIdx.x;
  if (t >= total) return;
  int c  = t % C;
  int hw = (t / C) % HW;
  int b  = t / (C * HW);
  Xt[t] = (_Float16)X[((long)b * C + c) * HW + hw];
}

// ---------------------------------------------------------------------------
// 1x1 conv as WMMA GEMM, Cin = 128 fixed. Y[b,o,hw] = sum_c W[o,c]*X[c,col].
// One wave: 16(out-ch) x 16(pixel) fp32 tile, 4 K-steps, all loads b128.
// block = 32x4, grid = (ceil(Ntot/64), Cout/16).
// ---------------------------------------------------------------------------
__global__ void gemm_conv1_k(const _Float16* __restrict__ Xt, // [Ntot][128]
                             const _Float16* __restrict__ Wf,
                             float* __restrict__ Y,
                             int Cout, int HW, int Ntot) {
  const int lane = threadIdx.x;
  const int half = lane >> 4, lm = lane & 15;
  const int tile = blockIdx.x * blockDim.y + threadIdx.y;
  int col = tile * 16 + lm;
  const bool valid = col < Ntot;
  if (!valid) col = Ntot - 1; // clamp: safe address, result discarded
  const int ot = blockIdx.y;
  const _Float16* xp = Xt + (long)col * 128 + (half << 3);
  const _Float16* wp = Wf + (((long)ot * 4) * 32 + lane) * 16;

  v8f acc = {};
#pragma unroll
  for (int kt = 0; kt < 4; ++kt) {
    v16h Af = *(const v16h*)(wp + kt * 32 * 16);
    v16h Bf = frag16(xp + kt * 32);
    acc = __builtin_amdgcn_wmma_f32_16x16x32_f16(false, Af, false, Bf,
                                                 (short)0, acc, false, false);
  }
  if (!valid) return;
  int b = col / HW, hw = col % HW;
#pragma unroll
  for (int r = 0; r < 8; ++r) {
    int o = ot * 16 + r + (half << 3);
    Y[((long)b * Cout + o) * HW + hw] = acc[r];
  }
}

// ---------------------------------------------------------------------------
// Depthwise 3x3 (cross-correlation, pad 1), optional bias and residual add.
// ---------------------------------------------------------------------------
__global__ void dwconv_k(const float* __restrict__ X, const float* __restrict__ W3,
                         const float* __restrict__ Bias, float* __restrict__ Y,
                         int C, int H, int Wd, int total, int addres) {
  int t = blockIdx.x * blockDim.x + threadIdx.x;
  if (t >= total) return;
  int x = t % Wd;
  int y = (t / Wd) % H;
  int c = (t / (Wd * H)) % C;
  int b = t / (Wd * H * C);
  const float* Xc = X + ((long)(b * C + c)) * H * Wd;
  const float* wk = W3 + c * 9;
  float acc = Bias ? Bias[c] : 0.f;
#pragma unroll
  for (int dy = -1; dy <= 1; ++dy) {
#pragma unroll
    for (int dx = -1; dx <= 1; ++dx) {
      int yy = y + dy, xx = x + dx;
      if (yy >= 0 && yy < H && xx >= 0 && xx < Wd)
        acc += wk[(dy + 1) * 3 + (dx + 1)] * Xc[yy * Wd + xx];
    }
  }
  if (addres) acc += Xc[y * Wd + x];
  Y[t] = acc;
}

// ---------------------------------------------------------------------------
// Window/head gather packs (f16, fragment-friendly):
//   Qh[bn][h][qpos][cc]  (qpos 0..63, cc 0..31)
//   Kh[bn][h][key][cc]   (key 0..1023)
//   Vh[bn][h][cc][key]
// ---------------------------------------------------------------------------
__global__ void pack_q_k(const float* __restrict__ Q, _Float16* __restrict__ Qh,
                         int total) {
  int t = blockIdx.x * blockDim.x + threadIdx.x;
  if (t >= total) return;
  int cc   = t & 31;
  int qpos = (t >> 5) & 63;
  int h    = (t >> 11) & 3;
  int bn   = t >> 13;
  int b = bn / 100, n = bn % 100;
  int ys = 6 * (n / 10) + (qpos >> 3);
  int xs = 6 * (n % 10) + (qpos & 7);
  Qh[t] = (_Float16)Q[(((long)b * 128 + h * 32 + cc) * 62 + ys) * 62 + xs];
}

__global__ void pack_kcc_k(const float* __restrict__ KV, _Float16* __restrict__ Kh,
                           int total) {
  int t = blockIdx.x * blockDim.x + threadIdx.x;
  if (t >= total) return;
  int cc  = t & 31;
  int key = (t >> 5) & 1023;
  int h   = (t >> 15) & 3;
  int bn  = t >> 17;
  int b = bn / 100, n = bn % 100;
  int kr = 24 * (n / 10) + (key >> 5);
  int kc = 24 * (n % 10) + (key & 31);
  Kh[t] = (_Float16)KV[((long)b * 256 + h * 32 + cc) * 61504 + kr * 248 + kc];
}

__global__ void pack_v_k(const float* __restrict__ KV, _Float16* __restrict__ Vh,
                         int total) {
  int t = blockIdx.x * blockDim.x + threadIdx.x;
  if (t >= total) return;
  int key = t & 1023;
  int cc  = (t >> 10) & 31;
  int h   = (t >> 15) & 3;
  int bn  = t >> 17;
  int b = bn / 100, n = bn % 100;
  int kr = 24 * (n / 10) + (key >> 5);
  int kc = 24 * (n % 10) + (key & 31);
  Vh[t] = (_Float16)KV[((long)b * 256 + 128 + h * 32 + cc) * 61504 + kr * 248 + kc];
}

// ---------------------------------------------------------------------------
// Flash-style windowed cross-attention on packed f16 operands.
// grid.x = b*100*4, block = 32x4; wave w owns q rows [16w,16w+16).
// ---------------------------------------------------------------------------
__global__ void attn_k(const _Float16* __restrict__ Qh,
                       const _Float16* __restrict__ Kh,
                       const _Float16* __restrict__ Vh,
                       float* __restrict__ OutWin) {
  const int lane = threadIdx.x, w = threadIdx.y;
  const int half = lane >> 4, lm = lane & 15;
  const int h   = blockIdx.x & 3;
  const int bn  = blockIdx.x >> 2;
  const int bnh = blockIdx.x;
  const float scale = 0.17677669529663687f; // 32^-0.5

  __shared__ __align__(16) _Float16 Pl[4][16][32];

  // Q A-fragment: row = q token (16w+lm), K = 32 head channels.
  const int qpos = 16 * w + lm;
  v16h Aq = frag16(Qh + (((long)bnh) * 64 + qpos) * 32 + (half << 3));

  v8f m_i, l_i, o0 = {}, o1 = {};
#pragma unroll
  for (int r = 0; r < 8; ++r) { m_i[r] = -3.0e38f; l_i[r] = 0.f; }

  const _Float16* kbase = Kh + (long)bnh * 1024 * 32;
  const _Float16* vbase = Vh + (long)bnh * 32 * 1024;

  for (int kb = 0; kb < 32; ++kb) {
    // ---- scores for 32 keys (two 16-key WMMA tiles) ----
    v8f z = {};
    v16h Bk0 = frag16(kbase + ((long)(kb * 32 + lm)) * 32 + (half << 3));
    v16h Bk1 = frag16(kbase + ((long)(kb * 32 + 16 + lm)) * 32 + (half << 3));
    v8f s0 = __builtin_amdgcn_wmma_f32_16x16x32_f16(false, Aq, false, Bk0,
                                                    (short)0, z, false, false);
    v8f s1 = __builtin_amdgcn_wmma_f32_16x16x32_f16(false, Aq, false, Bk1,
                                                    (short)0, z, false, false);

    // ---- online softmax over this 32-key block ----
    v8f mx;
#pragma unroll
    for (int r = 0; r < 8; ++r) {
      s0[r] *= scale; s1[r] *= scale;
      mx[r] = fmaxf(s0[r], s1[r]);
    }
#pragma unroll
    for (int msk = 1; msk <= 8; msk <<= 1)
#pragma unroll
      for (int r = 0; r < 8; ++r)
        mx[r] = fmaxf(mx[r], __shfl_xor(mx[r], msk, 32));

    v8f rs;
#pragma unroll
    for (int r = 0; r < 8; ++r) {
      float mnew = fmaxf(m_i[r], mx[r]);
      float cf = __expf(m_i[r] - mnew);
      s0[r] = __expf(s0[r] - mnew);
      s1[r] = __expf(s1[r] - mnew);
      rs[r] = s0[r] + s1[r];
      l_i[r] *= cf;
      m_i[r] = mnew;
      o0[r] *= cf; o1[r] *= cf;
    }
#pragma unroll
    for (int msk = 1; msk <= 8; msk <<= 1)
#pragma unroll
      for (int r = 0; r < 8; ++r)
        rs[r] += __shfl_xor(rs[r], msk, 32);
#pragma unroll
    for (int r = 0; r < 8; ++r) l_i[r] += rs[r];

    // ---- restage P (C/D layout) -> A-fragment via per-wave LDS ----
#pragma unroll
    for (int r = 0; r < 8; ++r) {
      Pl[w][r + (half << 3)][lm]      = (_Float16)s0[r];
      Pl[w][r + (half << 3)][16 + lm] = (_Float16)s1[r];
    }
    __syncthreads();
    v16h Ap = frag16(&Pl[w][lm][half << 3]);
    __syncthreads();

    // ---- O += P @ V (two 16-channel tiles) ----
    v16h Bv0 = frag16(vbase + ((long)lm) * 1024 + kb * 32 + (half << 3));
    v16h Bv1 = frag16(vbase + ((long)(16 + lm)) * 1024 + kb * 32 + (half << 3));
    o0 = __builtin_amdgcn_wmma_f32_16x16x32_f16(false, Ap, false, Bv0,
                                                (short)0, o0, false, false);
    o1 = __builtin_amdgcn_wmma_f32_16x16x32_f16(false, Ap, false, Bv1,
                                                (short)0, o1, false, false);
  }

  // ---- normalize, write window output: (b, n, c=h*32+ch, qpos) ----
  const long obase = ((long)bn * 128 + h * 32) * 64;
#pragma unroll
  for (int r = 0; r < 8; ++r) {
    int qp = 16 * w + r + (half << 3);
    float inv = 1.f / l_i[r];
    OutWin[obase + (long)lm * 64 + qp]        = o0[r] * inv;
    OutWin[obase + (long)(16 + lm) * 64 + qp] = o1[r] * inv;
  }
}

// ---------------------------------------------------------------------------
// Overlap-add reverse (gather form) with count normalization.
// ---------------------------------------------------------------------------
__global__ void reverse_k(const float* __restrict__ Win, float* __restrict__ Out,
                          int total) {
  int t = blockIdx.x * blockDim.x + threadIdx.x;
  if (t >= total) return;
  int x = t % 62;
  int y = (t / 62) % 62;
  int c = (t / (62 * 62)) % 128;
  int b = t / (62 * 62 * 128);
  float acc = 0.f;
  int cnt = 0;
  for (int wi = 0; wi < 10; ++wi) {
    int ry = y - 6 * wi;
    if (ry < 0 || ry >= 8) continue;
    for (int wj = 0; wj < 10; ++wj) {
      int rx = x - 6 * wj;
      if (rx < 0 || rx >= 8) continue;
      acc += Win[(((long)(b * 100 + wi * 10 + wj)) * 128 + c) * 64 + ry * 8 + rx];
      ++cnt;
    }
  }
  Out[t] = acc / (float)cnt;
}

// ---------------------------------------------------------------------------
extern "C" void kernel_launch(void* const* d_in, const int* in_sizes, int n_in,
                              void* d_out, int out_size, void* d_ws, size_t ws_size,
                              hipStream_t stream) {
  (void)in_sizes; (void)n_in; (void)out_size; (void)ws_size;
  const float* x     = (const float*)d_in[0]; // (2,128,248,248)
  const float* sp    = (const float*)d_in[1]; // (2,128,62,62)
  const float* wpos  = (const float*)d_in[2]; // (128,1,3,3)
  const float* bpos  = (const float*)d_in[3]; // (128,)
  const float* wq    = (const float*)d_in[4]; // (128,128)
  const float* wqdw  = (const float*)d_in[5]; // (128,1,3,3)
  const float* wkv   = (const float*)d_in[6]; // (256,128)
  const float* wkvdw = (const float*)d_in[7]; // (256,1,3,3)
  const float* wout  = (const float*)d_in[8]; // (128,128)
  float* out = (float*)d_out;                 // (2,128,62,62)

  const int NX   = 2 * 128 * 248 * 248; // 15,745,024
  const int NSP  = 2 * 128 * 62 * 62;   //    984,064
  const int NKV  = 2 * 256 * 248 * 248; // 31,490,048
  const int NWIN = 2 * 100 * 128 * 64;  //  1,638,400
  const int NQH  = 800 * 64 * 32;       //  1,638,400
  const int NKH  = 800 * 1024 * 32;     // 26,214,400

  char* ws = (char*)d_ws;
  size_t off = 0;
  auto alloc = [&](size_t bytes) -> void* {
    void* p = ws + off;
    off = (off + bytes + 255) & ~(size_t)255;
    return p;
  };
  float*    x2     = (float*)alloc((size_t)NX * 4);
  float*    sp2    = (float*)alloc((size_t)NSP * 4);
  float*    qpre   = (float*)alloc((size_t)NSP * 4);
  float*    q      = (float*)alloc((size_t)NSP * 4);
  float*    kvpre  = (float*)alloc((size_t)NKV * 4);
  float*    kv     = (float*)alloc((size_t)NKV * 4);
  _Float16* wq16   = (_Float16*)alloc(16384 * 2);
  _Float16* wkv16  = (_Float16*)alloc(32768 * 2);
  _Float16* wout16 = (_Float16*)alloc(16384 * 2);
  _Float16* x2t    = (_Float16*)alloc((size_t)NX * 2);
  _Float16* sp2t   = (_Float16*)alloc((size_t)NSP * 2);
  _Float16* revt   = (_Float16*)alloc((size_t)NSP * 2);
  _Float16* qh     = (_Float16*)alloc((size_t)NQH * 2);
  _Float16* kh     = (_Float16*)alloc((size_t)NKH * 2);
  _Float16* vh     = (_Float16*)alloc((size_t)NKH * 2);
  float*    owin   = (float*)alloc((size_t)NWIN * 4);
  float*    rev    = (float*)alloc((size_t)NSP * 4);

  // Pack weights into WMMA A-fragment order (f16).
  packw_k<<<(16384 + 255) / 256, 256, 0, stream>>>(wq, wq16, 128, 128);
  packw_k<<<(32768 + 255) / 256, 256, 0, stream>>>(wkv, wkv16, 256, 128);
  packw_k<<<(16384 + 255) / 256, 256, 0, stream>>>(wout, wout16, 128, 128);

  // Positional depthwise conv + residual.
  dwconv_k<<<(NX + 255) / 256, 256, 0, stream>>>(x, wpos, bpos, x2,
                                                 128, 248, 248, NX, 1);
  dwconv_k<<<(NSP + 255) / 256, 256, 0, stream>>>(sp, wpos, bpos, sp2,
                                                  128, 62, 62, NSP, 1);

  // q = dw3(1x1(sp2))
  pack_act_k<<<(NSP + 255) / 256, 256, 0, stream>>>(sp2, sp2t, 128, 3844, NSP);
  {
    dim3 blk(32, 4), g((7688 + 63) / 64, 8);
    gemm_conv1_k<<<g, blk, 0, stream>>>(sp2t, wq16, qpre, 128, 3844, 7688);
  }
  dwconv_k<<<(NSP + 255) / 256, 256, 0, stream>>>(qpre, wqdw, nullptr, q,
                                                  128, 62, 62, NSP, 0);

  // kv = dw3(1x1(x2))
  pack_act_k<<<(NX + 255) / 256, 256, 0, stream>>>(x2, x2t, 128, 61504, NX);
  {
    dim3 blk(32, 4), g((123008 + 63) / 64, 16);
    gemm_conv1_k<<<g, blk, 0, stream>>>(x2t, wkv16, kvpre, 256, 61504, 123008);
  }
  dwconv_k<<<(NKV + 255) / 256, 256, 0, stream>>>(kvpre, wkvdw, nullptr, kv,
                                                  256, 248, 248, NKV, 0);

  // Gather windows/heads into fragment-friendly f16 layouts.
  pack_q_k<<<(NQH + 255) / 256, 256, 0, stream>>>(q, qh, NQH);
  pack_kcc_k<<<(NKH + 255) / 256, 256, 0, stream>>>(kv, kh, NKH);
  pack_v_k<<<(NKH + 255) / 256, 256, 0, stream>>>(kv, vh, NKH);

  // Windowed cross-attention (flash style, WMMA).
  attn_k<<<2 * 100 * 4, dim3(32, 4), 0, stream>>>(qh, kh, vh, owin);

  // Overlap-add reverse with count normalization.
  reverse_k<<<(NSP + 255) / 256, 256, 0, stream>>>(owin, rev, NSP);

  // Final 1x1 projection -> d_out.
  pack_act_k<<<(NSP + 255) / 256, 256, 0, stream>>>(rev, revt, 128, 3844, NSP);
  {
    dim3 blk(32, 4), g((7688 + 63) / 64, 8);
    gemm_conv1_k<<<g, blk, 0, stream>>>(revt, wout16, out, 128, 3844, 7688);
  }
}
